// PredictCell_23356032155663
// MI455X (gfx1250) — matrix operand order repaired and checked by reading, not compiled, not run on hardware
//
#include <hip/hip_runtime.h>
#include <hip/hip_bf16.h>
#include <stdint.h>
#include <math.h>

typedef __bf16 bf16;
typedef __attribute__((ext_vector_type(16))) __bf16 v16bf;
typedef __attribute__((ext_vector_type(8)))  float  v8f;

#define BATCH 2
#define CDIM  288
#define TDIM  8
#define PHYS  49
#define NHEAD 8
#define HEADD 36
#define HPAD  64
#define PTOK  1024   // 32*32
#define QTOK  8192   // 8*32*32

// ---------------------------------------------------------------------------
// WMMA fragment helpers (gfx1250 wave32; k contiguous per lane half, same order for A and B)
// All loads are UNCONDITIONAL: callers guarantee in-bounds (padded) storage,
// so the inner loops carry no EXEC-mask manipulation (WMMA needs EXEC=~0
// anyway; predication in the K-loop was pure overhead).
// ---------------------------------------------------------------------------
// A 16x32 bf16: lane L holds row (L&15); VGPR0-7 = k in [koff, koff+16),
// koff = (L<16 ? 0 : 16).
__device__ __forceinline__ v16bf load_a_frag(const bf16* A, int lda, int mbase,
                                             int kbase, int lane) {
  int row  = mbase + (lane & 15);
  int koff = kbase + ((lane >> 4) << 4);
  const bf16* p = A + (size_t)row * lda + koff;
  union { v16bf v; uint4 q[2]; } u;
  u.q[0] = *reinterpret_cast<const uint4*>(p);
  u.q[1] = *reinterpret_cast<const uint4*>(p + 8);
  return u.v;
}

// B 32x16 bf16 (K x N): lane L holds column (L&15); VGPR0-7 = k in
// [koff, koff+16), koff = (L<16 ? 0 : 16). W is stored [N][K] row-major,
// so column n's K-run is a contiguous 32-byte read.
__device__ __forceinline__ v16bf load_b_frag(const bf16* W, int ldw, int nbase,
                                             int kbase, int lane) {
  int n    = nbase + (lane & 15);
  int koff = kbase + ((lane >> 4) << 4);
  const bf16* p = W + (size_t)n * ldw + koff;
  union { v16bf v; uint4 q[2]; } u;
  u.q[0] = *reinterpret_cast<const uint4*>(p);
  u.q[1] = *reinterpret_cast<const uint4*>(p + 8);
  return u.v;
}

__device__ __forceinline__ v8f wmma_bf16(v16bf a, v16bf b, v8f c) {
  return __builtin_amdgcn_wmma_f32_16x16x32_bf16(false, a, false, b,
                                                 (short)0, c, false, false);
}

// ---------------------------------------------------------------------------
// Positional encodings (computed on the fly)
// ---------------------------------------------------------------------------
#define LN1E4 9.210340371976184f

__device__ __forceinline__ float pe2d_val(int c, int y, int x) {
  int   j   = (c < 144) ? c : c - 144;
  float pos = (c < 144) ? (float)x : (float)y;
  float dv  = __expf((float)(j & ~1) * (-LN1E4 / 144.0f));
  float ang = pos * dv;
  return (j & 1) ? __cosf(ang) : __sinf(ang);
}

__device__ __forceinline__ float pe3d_val(int c, int t, int y, int x) {
  int   seg = c / 96;
  int   j   = c - seg * 96;
  float pos = (seg == 0) ? (float)t : ((seg == 1) ? (float)y : (float)x);
  float dv  = __expf((float)(j & ~1) * (-LN1E4 / 96.0f));
  float ang = pos * dv;
  return (j & 1) ? __cosf(ang) : __sinf(ang);
}

// ---------------------------------------------------------------------------
// Prep kernels
// ---------------------------------------------------------------------------
__global__ void fill_zero_kernel(uint32_t* p, int n) {
  int i = blockIdx.x * blockDim.x + threadIdx.x;
  if (i < n) p[i] = 0u;
}

// dst is [rows_pad][ldst]; src is [rows][cols]; out-of-range entries -> 0.
__global__ void cvt_weight_kernel(const float* __restrict__ src, bf16* __restrict__ dst,
                                  int rows, int cols, int rows_pad, int ldst) {
  int idx = blockIdx.x * blockDim.x + threadIdx.x;
  if (idx >= rows_pad * ldst) return;
  int r = idx / ldst, c = idx - r * ldst;
  dst[idx] = (r < rows && c < cols) ? (bf16)src[(size_t)r * cols + c] : (bf16)0.0f;
}

// q_src[b*1024+p][c] = new_update + pe2d  (token-major bf16)
__global__ void build_q_kernel(const float* __restrict__ nu, bf16* __restrict__ qsrc) {
  int idx = blockIdx.x * blockDim.x + threadIdx.x;
  if (idx >= BATCH * PTOK * CDIM) return;
  int c = idx % CDIM, m = idx / CDIM;
  int b = m >> 10, p = m & 1023, y = p >> 5, x = p & 31;
  float v = nu[((size_t)(b * CDIM + c)) * PTOK + p] + pe2d_val(c, y, x);
  qsrc[idx] = (bf16)v;
}

// k_src = prev_x + pe3d ; v_src = prev_x   (token-major bf16)
__global__ void build_kv_kernel(const float* __restrict__ px,
                                bf16* __restrict__ ksrc, bf16* __restrict__ vsrc) {
  int idx = blockIdx.x * blockDim.x + threadIdx.x;
  if (idx >= BATCH * QTOK * CDIM) return;
  int c = idx % CDIM, m = idx / CDIM;
  int b = m >> 13, q = m & 8191;
  int t = q >> 10, p = q & 1023, y = p >> 5, x = p & 31;
  float v = px[(((size_t)(b * CDIM + c)) * TDIM + t) * PTOK + p];
  ksrc[idx] = (bf16)(v + pe3d_val(c, t, y, x));
  vsrc[idx] = (bf16)v;
}

// ---------------------------------------------------------------------------
// Generic bf16 WMMA GEMM:  D[m][n] = scale * sum_k A[m][k]*W[n][k] + bias[n]
// W must be padded to ceil(N/16)*16 rows (zero rows); inner loop is
// branch-free: 4x global_load_b128 + v_wmma per K-step.
// ---------------------------------------------------------------------------
enum { MODE_QH = 1, MODE_KH = 2, MODE_VT = 3, MODE_BCHW = 4, MODE_FINAL = 5 };

__global__ void gemm_bf16_kernel(const bf16* __restrict__ A, const bf16* __restrict__ W,
                                 const float* __restrict__ bias, int M, int N, int K,
                                 float scale, int mode, float* __restrict__ outf,
                                 bf16* __restrict__ outbf, const float* __restrict__ add) {
  int lane  = threadIdx.x & 31;
  int mtile = blockIdx.y * (blockDim.x >> 5) + (threadIdx.x >> 5);
  int mbase = mtile * 16;
  int nbase = blockIdx.x * 16;
  if (mbase >= M) return;

  v8f acc = {};
  for (int k = 0; k < K; k += 32) {
    v16bf a = load_a_frag(A, K, mbase, k, lane);
    v16bf b = load_b_frag(W, K, nbase, k, lane);
    acc = wmma_bf16(a, b, acc);
  }

  int n = nbase + (lane & 15);
  if (n >= N) return;
  float bval = bias ? bias[n] : 0.0f;
  int mh = (lane >> 4) << 3;   // C/D layout: lanes>=16 hold rows 8..15
#pragma unroll
  for (int r = 0; r < 8; ++r) {
    int m = mbase + mh + r;
    float v = acc[r] * scale + bval;
    switch (mode) {
      case MODE_QH: {  // [B,NH,1024,64] bf16, pad zeros preserved
        int b_ = m >> 10, p = m & 1023, hd = n / HEADD, d = n - hd * HEADD;
        outbf[(((size_t)(b_ * NHEAD + hd) * PTOK + p) * HPAD) + d] = (bf16)v;
      } break;
      case MODE_KH: {  // [B,NH,8192,64] bf16
        int b_ = m >> 13, q = m & 8191, hd = n / HEADD, d = n - hd * HEADD;
        outbf[(((size_t)(b_ * NHEAD + hd) * QTOK + q) * HPAD) + d] = (bf16)v;
      } break;
      case MODE_VT: {  // V transposed: [B,NH,64,8192] bf16
        int b_ = m >> 13, q = m & 8191, hd = n / HEADD, d = n - hd * HEADD;
        outbf[(((size_t)(b_ * NHEAD + hd) * HPAD + d) * QTOK) + q] = (bf16)v;
      } break;
      case MODE_BCHW: { // [B,N,1024] f32
        int b_ = m >> 10, p = m & 1023;
        outf[((size_t)b_ * N + n) * PTOK + p] = v;
      } break;
      case MODE_FINAL: { // d_out = v + new_update, [B,288,1024] f32
        int b_ = m >> 10, p = m & 1023;
        size_t idx = ((size_t)b_ * CDIM + n) * PTOK + p;
        outf[idx] = v + add[idx];
      } break;
    }
  }
}

// ---------------------------------------------------------------------------
// Flash attention: one wave per (b, head, 16-query tile), 32-key tiles,
// online softmax, P routed through LDS to re-layout C->A fragments.
// All fragment loads are in-bounds by construction (keys < 8192, dims < 64
// with rows 36..63 zero-padded), so no predication anywhere in the loop.
// ---------------------------------------------------------------------------
#define FA_WAVES 4

__global__ void flash_attn_kernel(const bf16* __restrict__ Qh, const bf16* __restrict__ Kh,
                                  const bf16* __restrict__ Vt, bf16* __restrict__ attn) {
  __shared__ bf16 lds[FA_WAVES][16 * 32];
  int lane = threadIdx.x & 31;
  int wave = threadIdx.x >> 5;
  int gid  = blockIdx.x * FA_WAVES + wave;       // 0 .. B*NH*64-1
  int ptile = gid & 63;
  int hd    = (gid >> 6) & (NHEAD - 1);
  int b     = gid >> 9;
  const bf16* Qbase = Qh + (((size_t)(b * NHEAD + hd) * PTOK) + ptile * 16) * HPAD;
  const bf16* Kbase = Kh + ((size_t)(b * NHEAD + hd) * QTOK) * HPAD;
  const bf16* Vbase = Vt + ((size_t)(b * NHEAD + hd) * HPAD) * QTOK;
  bf16* pbuf = lds[wave];

  v16bf qf0 = load_a_frag(Qbase, HPAD, 0, 0, lane);
  v16bf qf1 = load_a_frag(Qbase, HPAD, 0, 32, lane);

  float rowm[8], rowl[8];
#pragma unroll
  for (int r = 0; r < 8; ++r) { rowm[r] = -3.0e38f; rowl[r] = 0.0f; }
  v8f oacc0 = {}, oacc1 = {}, oacc2 = {};

  int col = lane & 15;
  int rowoff = (lane >> 4) << 3;

  for (int kt = 0; kt < QTOK; kt += 32) {
    // S = Q (16x64) @ K^T (64x32)  -> two 16x16 N-subtiles, each K=64 in 2 steps
    v8f s0 = {}, s1 = {};
    s0 = wmma_bf16(qf0, load_b_frag(Kbase, HPAD, kt,      0,  lane), s0);
    s0 = wmma_bf16(qf1, load_b_frag(Kbase, HPAD, kt,      32, lane), s0);
    s1 = wmma_bf16(qf0, load_b_frag(Kbase, HPAD, kt + 16, 0,  lane), s1);
    s1 = wmma_bf16(qf1, load_b_frag(Kbase, HPAD, kt + 16, 32, lane), s1);

    // Online softmax over this 32-key slab; row stats via width-16 shuffles.
#pragma unroll
    for (int r = 0; r < 8; ++r) {
      float mx = fmaxf(s0[r], s1[r]);
#pragma unroll
      for (int off = 8; off; off >>= 1) mx = fmaxf(mx, __shfl_xor(mx, off, 16));
      float nm = fmaxf(rowm[r], mx);
      float p0 = __expf(s0[r] - nm);
      float p1 = __expf(s1[r] - nm);
      float sum = p0 + p1;
#pragma unroll
      for (int off = 8; off; off >>= 1) sum += __shfl_xor(sum, off, 16);
      float sc = __expf(rowm[r] - nm);
      rowl[r] = rowl[r] * sc + sum;
      rowm[r] = nm;
      oacc0[r] *= sc; oacc1[r] *= sc; oacc2[r] *= sc;
      // stage P (16x32 row-major bf16) into LDS for A-layout reload
      int row = rowoff + r;
      pbuf[row * 32 + col]      = (bf16)p0;
      pbuf[row * 32 + 16 + col] = (bf16)p1;
    }
    asm volatile("s_wait_dscnt 0x0" ::: "memory");
    v16bf pf = load_a_frag(pbuf, 32, 0, 0, lane);

    // O += P (16x32) @ V (32x48), 3 N-subtiles (head dim padded; 36..47 are 0)
    oacc0 = wmma_bf16(pf, load_b_frag(Vbase, QTOK, 0,  kt, lane), oacc0);
    oacc1 = wmma_bf16(pf, load_b_frag(Vbase, QTOK, 16, kt, lane), oacc1);
    oacc2 = wmma_bf16(pf, load_b_frag(Vbase, QTOK, 32, kt, lane), oacc2);
  }

  // Epilogue: normalize by l and scatter to token-major [B*1024][288] bf16
#pragma unroll
  for (int r = 0; r < 8; ++r) {
    int m = ptile * 16 + rowoff + r;
    float inv = 1.0f / rowl[r];
    bf16* dst = attn + ((size_t)b * PTOK + m) * CDIM + hd * HEADD;
    dst[col]      = (bf16)(oacc0[r] * inv);
    dst[16 + col] = (bf16)(oacc1[r] * inv);
    if (32 + col < HEADD) dst[32 + col] = (bf16)(oacc2[r] * inv);
  }
}

// ---------------------------------------------------------------------------
// conv3x3 SAME (288 -> 49), direct
// ---------------------------------------------------------------------------
__global__ void conv3x3_kernel(const float* __restrict__ x, const float* __restrict__ Fw,
                               const float* __restrict__ Fb, float* __restrict__ out) {
  int idx = blockIdx.x * blockDim.x + threadIdx.x;
  if (idx >= BATCH * PHYS * PTOK) return;
  int p = idx & 1023;
  int o = (idx >> 10) % PHYS;
  int b = idx / (PHYS * PTOK);
  int y = p >> 5, xw = p & 31;
  float s = Fb[o];
  for (int c = 0; c < CDIM; ++c) {
    const float* xp = x + ((size_t)b * CDIM + c) * PTOK;
    const float* wp = Fw + ((size_t)o * CDIM + c) * 9;
#pragma unroll
    for (int dy = -1; dy <= 1; ++dy) {
      int yy = y + dy;
      if (yy < 0 || yy > 31) continue;
#pragma unroll
      for (int dx = -1; dx <= 1; ++dx) {
        int xx = xw + dx;
        if (xx < 0 || xx > 31) continue;
        s += xp[yy * 32 + xx] * wp[(dy + 1) * 3 + (dx + 1)];
      }
    }
  }
  out[idx] = s;
}

// ---------------------------------------------------------------------------
// GroupNorm over [B, CC, 1024]; one block per (b, group).
// transposed=0: f32 same layout. transposed=1: bf16 token-major [B*1024][ldout].
// ---------------------------------------------------------------------------
__global__ void groupnorm_kernel(const float* __restrict__ x, const float* __restrict__ gamma,
                                 const float* __restrict__ beta, int CC, int G,
                                 int transposed, float* __restrict__ outf,
                                 bf16* __restrict__ outbf, int ldout) {
  int b = blockIdx.x / G;
  int g = blockIdx.x % G;
  int cpg = CC / G;
  size_t base = ((size_t)b * CC + (size_t)g * cpg) * PTOK;
  int n = cpg * PTOK;
  float s = 0.0f, s2 = 0.0f;
  for (int i = threadIdx.x; i < n; i += blockDim.x) {
    float v = x[base + i];
    s += v; s2 += v * v;
  }
  __shared__ float sh0[256], sh1[256];
  sh0[threadIdx.x] = s; sh1[threadIdx.x] = s2;
  __syncthreads();
  for (int st = 128; st > 0; st >>= 1) {
    if ((int)threadIdx.x < st) {
      sh0[threadIdx.x] += sh0[threadIdx.x + st];
      sh1[threadIdx.x] += sh1[threadIdx.x + st];
    }
    __syncthreads();
  }
  float mu  = sh0[0] / (float)n;
  float var = sh1[0] / (float)n - mu * mu;
  float rs  = rsqrtf(var + 1e-5f);
  for (int i = threadIdx.x; i < n; i += blockDim.x) {
    int c = g * cpg + i / PTOK;
    int p = i & (PTOK - 1);
    float v = (x[base + i] - mu) * rs * gamma[c] + beta[c];
    if (!transposed) outf[base + i] = v;
    else outbf[((size_t)b * PTOK + p) * ldout + c] = (bf16)v;
  }
}

__global__ void mul_kernel(const float* __restrict__ a, const float* __restrict__ b,
                           float* __restrict__ o, int n) {
  int i = blockIdx.x * blockDim.x + threadIdx.x;
  if (i < n) o[i] = a[i] * b[i];
}

// ---------------------------------------------------------------------------
// Launch
// ---------------------------------------------------------------------------
extern "C" void kernel_launch(void* const* d_in, const int* in_sizes, int n_in,
                              void* d_out, int out_size, void* d_ws, size_t ws_size,
                              hipStream_t stream) {
  const float* prev_x     = (const float*)d_in[1];
  const float* new_update = (const float*)d_in[2];
  const float* Fw   = (const float*)d_in[3];
  const float* Fb   = (const float*)d_in[4];
  const float* gn_g = (const float*)d_in[5];
  const float* gn_b = (const float*)d_in[6];
  const float* l1gg = (const float*)d_in[7];
  const float* l1gb = (const float*)d_in[8];
  const float* l1w  = (const float*)d_in[9];
  const float* l1b  = (const float*)d_in[10];
  const float* l2gg = (const float*)d_in[11];
  const float* l2gb = (const float*)d_in[12];
  const float* l2w  = (const float*)d_in[13];
  const float* l2b  = (const float*)d_in[14];
  const float* Wq = (const float*)d_in[15]; const float* bq = (const float*)d_in[16];
  const float* Wk = (const float*)d_in[17]; const float* bk = (const float*)d_in[18];
  const float* Wv = (const float*)d_in[19]; const float* bv = (const float*)d_in[20];
  const float* Wo = (const float*)d_in[21]; const float* bo = (const float*)d_in[22];

  size_t off = 0;
  auto alloc = [&](size_t bytes) -> void* {
    void* p = (char*)d_ws + off;
    off += (bytes + 255) & ~(size_t)255;
    return p;
  };
  bf16*  q_src   = (bf16*)alloc((size_t)BATCH * PTOK * CDIM * 2);
  bf16*  k_src   = (bf16*)alloc((size_t)BATCH * QTOK * CDIM * 2);
  bf16*  v_src   = (bf16*)alloc((size_t)BATCH * QTOK * CDIM * 2);
  bf16*  wq_bf   = (bf16*)alloc((size_t)CDIM * CDIM * 2);
  bf16*  wk_bf   = (bf16*)alloc((size_t)CDIM * CDIM * 2);
  bf16*  wv_bf   = (bf16*)alloc((size_t)CDIM * CDIM * 2);
  bf16*  wo_bf   = (bf16*)alloc((size_t)CDIM * CDIM * 2);
  bf16*  l1w_bf  = (bf16*)alloc((size_t)64 * CDIM * 2);    // rows padded 49->64
  bf16*  l2w_bf  = (bf16*)alloc((size_t)CDIM * 64 * 2);    // cols padded 49->64
  bf16*  Qhb     = (bf16*)alloc((size_t)BATCH * NHEAD * PTOK * HPAD * 2);
  bf16*  Khb     = (bf16*)alloc((size_t)BATCH * NHEAD * QTOK * HPAD * 2);
  bf16*  Vtb     = (bf16*)alloc((size_t)BATCH * NHEAD * HPAD * QTOK * 2);
  bf16*  attn_bf = (bf16*)alloc((size_t)BATCH * PTOK * CDIM * 2);
  float* coef    = (float*)alloc((size_t)BATCH * CDIM * PTOK * 4);
  bf16*  coefn   = (bf16*)alloc((size_t)BATCH * PTOK * CDIM * 2);
  float* coef49  = (float*)alloc((size_t)BATCH * PHYS * PTOK * 4);
  float* hg      = (float*)alloc((size_t)BATCH * PHYS * PTOK * 4);
  float* hgn     = (float*)alloc((size_t)BATCH * PHYS * PTOK * 4);
  float* prod    = (float*)alloc((size_t)BATCH * PHYS * PTOK * 4);
  bf16*  prodn   = (bf16*)alloc((size_t)BATCH * PTOK * 64 * 2);

  // zero pad-sensitive buffers (Qh/Kh/Vt head-dim pads, prodn K-pad)
  {
    int nqh = BATCH * NHEAD * PTOK * HPAD / 2;
    int nkh = BATCH * NHEAD * QTOK * HPAD / 2;
    int npd = BATCH * PTOK * 64 / 2;
    fill_zero_kernel<<<(nqh + 255) / 256, 256, 0, stream>>>((uint32_t*)Qhb, nqh);
    fill_zero_kernel<<<(nkh + 255) / 256, 256, 0, stream>>>((uint32_t*)Khb, nkh);
    fill_zero_kernel<<<(nkh + 255) / 256, 256, 0, stream>>>((uint32_t*)Vtb, nkh);
    fill_zero_kernel<<<(npd + 255) / 256, 256, 0, stream>>>((uint32_t*)prodn, npd);
  }

  // weights -> bf16 (row/col padded so GEMM B-loads need no bounds checks)
  cvt_weight_kernel<<<324, 256, 0, stream>>>(Wq, wq_bf, CDIM, CDIM, CDIM, CDIM);
  cvt_weight_kernel<<<324, 256, 0, stream>>>(Wk, wk_bf, CDIM, CDIM, CDIM, CDIM);
  cvt_weight_kernel<<<324, 256, 0, stream>>>(Wv, wv_bf, CDIM, CDIM, CDIM, CDIM);
  cvt_weight_kernel<<<324, 256, 0, stream>>>(Wo, wo_bf, CDIM, CDIM, CDIM, CDIM);
  cvt_weight_kernel<<<72, 256, 0, stream>>>(l1w, l1w_bf, PHYS, CDIM, 64, CDIM);
  cvt_weight_kernel<<<72, 256, 0, stream>>>(l2w, l2w_bf, CDIM, PHYS, CDIM, 64);

  // token-major bf16 activations with positional encodings
  build_q_kernel<<<2304, 256, 0, stream>>>(new_update, q_src);
  build_kv_kernel<<<18432, 256, 0, stream>>>(prev_x, k_src, v_src);

  // h_grad = gn7(conv3x3(new_update))
  conv3x3_kernel<<<392, 256, 0, stream>>>(new_update, Fw, Fb, hg);
  groupnorm_kernel<<<BATCH * 7, 256, 0, stream>>>(hg, gn_g, gn_b, PHYS, 7, 0, hgn, nullptr, 0);

  // QKV projections (energy scale folded into Q)
  const float inv_sqrt = 0.05892556509887896f;  // 1/sqrt(NH*HD) = 1/sqrt(288)
  gemm_bf16_kernel<<<dim3(18, 32), 128, 0, stream>>>(q_src, wq_bf, bq, 2048, CDIM, CDIM,
                                                     inv_sqrt, MODE_QH, nullptr, Qhb, nullptr);
  gemm_bf16_kernel<<<dim3(18, 256), 128, 0, stream>>>(k_src, wk_bf, bk, 16384, CDIM, CDIM,
                                                      1.0f, MODE_KH, nullptr, Khb, nullptr);
  gemm_bf16_kernel<<<dim3(18, 256), 128, 0, stream>>>(v_src, wv_bf, bv, 16384, CDIM, CDIM,
                                                      1.0f, MODE_VT, nullptr, Vtb, nullptr);

  // streaming-softmax attention
  flash_attn_kernel<<<BATCH * NHEAD * 64 / FA_WAVES, FA_WAVES * 32, 0, stream>>>(
      Qhb, Khb, Vtb, attn_bf);

  // coef = Wo @ attn + bo ; gn8 ; lin1
  gemm_bf16_kernel<<<dim3(18, 32), 128, 0, stream>>>(attn_bf, wo_bf, bo, 2048, CDIM, CDIM,
                                                     1.0f, MODE_BCHW, coef, nullptr, nullptr);
  groupnorm_kernel<<<BATCH * 8, 256, 0, stream>>>(coef, l1gg, l1gb, CDIM, 8, 1,
                                                  nullptr, coefn, CDIM);
  gemm_bf16_kernel<<<dim3(4, 32), 128, 0, stream>>>(coefn, l1w_bf, l1b, 2048, PHYS, CDIM,
                                                    1.0f, MODE_BCHW, coef49, nullptr, nullptr);

  // new_predict = lin2(gn7(coef * h_grad)) + new_update
  mul_kernel<<<392, 256, 0, stream>>>(coef49, hgn, prod, BATCH * PHYS * PTOK);
  groupnorm_kernel<<<BATCH * 7, 256, 0, stream>>>(prod, l2gg, l2gb, PHYS, 7, 1,
                                                  nullptr, prodn, 64);
  gemm_bf16_kernel<<<dim3(18, 32), 128, 0, stream>>>(prodn, l2w_bf, l2b, 2048, CDIM, 64,
                                                     1.0f, MODE_FINAL, (float*)d_out,
                                                     nullptr, new_update);
}
